// Block_42399917146348
// MI455X (gfx1250) — compile-verified, matrix operand order checked by reading, not run on hardware
//
#include <hip/hip_runtime.h>
#include <hip/hip_bf16.h>
#include <math.h>

// ---------------- problem constants (match reference) ----------------
#define BB 2
#define SS 2048
#define HH 16
#define DHD 64
#define DD 1024
#define PFF 4096

typedef __attribute__((ext_vector_type(16))) _Float16 v16h;
typedef __attribute__((ext_vector_type(4)))  _Float16 v4h;
typedef __attribute__((ext_vector_type(8)))  float    v8f;

__device__ __forceinline__ v8f wmma_f16(v16h a, v16h b, v8f c) {
    return __builtin_amdgcn_wmma_f32_16x16x32_f16(false, a, false, b, (short)0, c, false, false);
}

// =====================================================================
// Fragment-order LDS staging (ISA 7.12.2 wave32 layouts):
//   element (m|n, k):  lane l = (mn & 15) + 16*((k>>3)&1)
//                      elem e = (k & 7) + 8*(k>>4)
// Each fragment's 16 halves stored contiguously: lds[sub*512 + l*16 + e].
// Staging is split into (a) global float4 loads into registers and
// (b) cvt + LDS store, so the pipelined kernels can overlap (a) with WMMA.
// =====================================================================

struct RegA { float4 f[4]; };   // 128 rows x 32 k tile, per-thread share
struct RegB { float4 f[2]; };   // 32 k x 64 n tile, per-thread share

// ---- A tile: global f32 row-major (ld=lda), 128x32 ----
__device__ __forceinline__ RegA gload_a(const float* __restrict__ A, int lda,
                                        int row_base, int k0, int tid) {
    RegA r;
#pragma unroll
    for (int i = 0; i < 4; ++i) {
        int gidx = i * 256 + tid;      // 0..1023
        int row  = gidx >> 3;          // 0..127
        int kg   = gidx & 7;           // k-group of 4
        r.f[i] = *(const float4*)(A + (size_t)(row_base + row) * lda + k0 + kg * 4);
    }
    return r;
}
__device__ __forceinline__ void sstore_a(_Float16* lA, const RegA& r, int tid) {
#pragma unroll
    for (int i = 0; i < 4; ++i) {
        int gidx = i * 256 + tid;
        int row  = gidx >> 3;
        int kg   = gidx & 7;
        int l    = (row & 15) + (((kg >> 1) & 1) << 4);
        int e0   = ((kg & 1) << 2) + ((kg >> 2) << 3);
        v4h h = { (_Float16)r.f[i].x, (_Float16)r.f[i].y,
                  (_Float16)r.f[i].z, (_Float16)r.f[i].w };
        *(v4h*)(lA + ((row >> 4) << 9) + l * 16 + e0) = h;   // ds_store_b64
    }
}

// ---- B tile from row-major [K,N] global (ld=ldb): 32x64 ----
__device__ __forceinline__ RegB gload_b(const float* __restrict__ Bm, int ldb,
                                        int k0, int n0, int tid) {
    RegB r;
#pragma unroll
    for (int i = 0; i < 2; ++i) {
        int gidx = i * 256 + tid;      // 0..511
        int k    = gidx >> 4;          // 0..31
        int ng   = gidx & 15;          // n-group of 4
        r.f[i] = *(const float4*)(Bm + (size_t)(k0 + k) * ldb + n0 + ng * 4);
    }
    return r;
}
__device__ __forceinline__ void sstore_b(_Float16* lB, const RegB& r, int tid) {
#pragma unroll
    for (int i = 0; i < 2; ++i) {
        int gidx = i * 256 + tid;
        int k    = gidx >> 4;
        int ng   = gidx & 15;
        int t    = ng >> 2;
        int e    = (k & 7) + ((k >> 4) << 3);
        int lb   = ((ng & 3) << 2) + (((k >> 3) & 1) << 4);
        _Float16* dst = lB + (t << 9) + lb * 16 + e;
        dst[0]  = (_Float16)r.f[i].x;
        dst[16] = (_Float16)r.f[i].y;
        dst[32] = (_Float16)r.f[i].z;
        dst[48] = (_Float16)r.f[i].w;
    }
}

// ---- B = X^T tile (logits): B(k,n) = X[(n0+n)*ldx + k0+k] ----
__device__ __forceinline__ RegB gload_bT(const float* __restrict__ X, int ldx,
                                         int k0, int n0, int tid) {
    RegB r;
#pragma unroll
    for (int i = 0; i < 2; ++i) {
        int gidx = i * 256 + tid;      // 0..511
        int n    = gidx >> 3;          // 0..63
        int kg   = gidx & 7;
        r.f[i] = *(const float4*)(X + (size_t)(n0 + n) * ldx + k0 + kg * 4);
    }
    return r;
}
__device__ __forceinline__ void sstore_bT(_Float16* lB, const RegB& r, int tid) {
#pragma unroll
    for (int i = 0; i < 2; ++i) {
        int gidx = i * 256 + tid;
        int n    = gidx >> 3;
        int kg   = gidx & 7;
        int t    = n >> 4;
        int l    = (n & 15) + (((kg >> 1) & 1) << 4);
        int e0   = ((kg & 1) << 2) + ((kg >> 2) << 3);
        v4h h = { (_Float16)r.f[i].x, (_Float16)r.f[i].y,
                  (_Float16)r.f[i].z, (_Float16)r.f[i].w };
        *(v4h*)(lB + (t << 9) + l * 16 + e0) = h;
    }
}

__device__ __forceinline__ v16h frag_a(const _Float16* lA, int wave, int lane) {
    return *(const v16h*)(lA + (wave << 9) + lane * 16);     // 2x ds_load_b128
}
__device__ __forceinline__ v16h frag_b(const _Float16* lB, int t, int lane) {
    return *(const v16h*)(lB + (t << 9) + lane * 16);
}

// Batched fragment loads + 4 WMMAs (one DS wait instead of four).
__device__ __forceinline__ void compute_step(const _Float16* lA, const _Float16* lB,
                                             int wave, int lane, v8f acc[4]) {
    v16h a  = frag_a(lA, wave, lane);
    v16h b0 = frag_b(lB, 0, lane);
    v16h b1 = frag_b(lB, 1, lane);
    v16h b2 = frag_b(lB, 2, lane);
    v16h b3 = frag_b(lB, 3, lane);
    acc[0] = wmma_f16(a, b0, acc[0]);
    acc[1] = wmma_f16(a, b1, acc[1]);
    acc[2] = wmma_f16(a, b2, acc[2]);
    acc[3] = wmma_f16(a, b3, acc[3]);
}

// ---------------- generic GEMM: C = act(A@Bm + bias [+ Res]) ----------------
// 256 threads = 8 waves; tile 128(M) x 64(N); K-step 32; ping-pong LDS pipeline.
template <int RELU, int RES>
__global__ __launch_bounds__(256)
void gemm_wmma_kernel(const float* __restrict__ A, const float* __restrict__ Bm,
                      const float* __restrict__ bias, const float* __restrict__ Res,
                      float* __restrict__ C, int M, int N, int K) {
    __shared__ _Float16 lA[2 * 128 * 32];
    __shared__ _Float16 lB[2 * 4 * 512];
    int tid  = threadIdx.x;
    int lane = tid & 31;
    int wave = tid >> 5;
    int rowB = blockIdx.y * 128;
    int n0   = blockIdx.x * 64;

    RegA ra = gload_a(A, K, rowB, 0, tid);
    RegB rb = gload_b(Bm, N, 0, n0, tid);
    sstore_a(lA, ra, tid);
    sstore_b(lB, rb, tid);

    v8f acc[4] = {};
    int cur = 0;
    for (int k0 = 0; k0 < K; k0 += 32) {
        __syncthreads();
        bool more = (k0 + 32) < K;
        if (more) {                                  // issue next tile's global loads
            ra = gload_a(A, K, rowB, k0 + 32, tid);
            rb = gload_b(Bm, N, k0 + 32, n0, tid);
        }
        compute_step(lA + cur * 4096, lB + cur * 2048, wave, lane, acc);  // overlap
        if (more) {                                  // drain loads into other buffer
            sstore_a(lA + (cur ^ 1) * 4096, ra, tid);
            sstore_b(lB + (cur ^ 1) * 2048, rb, tid);
            cur ^= 1;
        }
    }

    int g  = (lane >> 4) & 1;
    int nc = lane & 15;
    int row0 = rowB + wave * 16;
#pragma unroll
    for (int t = 0; t < 4; ++t) {
#pragma unroll
        for (int r = 0; r < 8; ++r) {
            int row = row0 + r + 8 * g;
            int col = n0 + t * 16 + nc;
            float v = acc[t][r] + bias[col];
            if (RES)  v += Res[(size_t)row * N + col];
            if (RELU) v = fmaxf(v, 0.0f);
            C[(size_t)row * N + col] = v;
        }
    }
}

// ---------------- attention logits: L = Q·K^T/8 + mask*(-1e9) ----------------
// K-loop is only DH/32 = 2 steps; output stream dominates -> simple scheme.
__global__ __launch_bounds__(256)
void attn_logits_kernel(const float* __restrict__ Q, const float* __restrict__ Km,
                        const int* __restrict__ mask, float* __restrict__ Lg) {
    __shared__ _Float16 lA[2 * 128 * 32];
    __shared__ _Float16 lB[2 * 4 * 512];
    int tid  = threadIdx.x;
    int lane = tid & 31;
    int wave = tid >> 5;
    int bh = blockIdx.z;
    int b = bh / HH, h = bh % HH;
    int qB = blockIdx.y * 128;
    int n0 = blockIdx.x * 64;

    const float* Qh = Q  + (size_t)b * SS * (HH * DHD) + h * DHD;
    const float* Kh = Km + (size_t)b * SS * (HH * DHD) + h * DHD;

    // stage both K-halves at once (two buffers), one barrier
    RegA ra0 = gload_a(Qh, HH * DHD, qB, 0, tid);
    RegB rb0 = gload_bT(Kh, HH * DHD, 0, n0, tid);
    RegA ra1 = gload_a(Qh, HH * DHD, qB, 32, tid);
    RegB rb1 = gload_bT(Kh, HH * DHD, 32, n0, tid);
    sstore_a (lA, ra0, tid);        sstore_bT(lB, rb0, tid);
    sstore_a (lA + 4096, ra1, tid); sstore_bT(lB + 2048, rb1, tid);
    __syncthreads();

    v8f acc[4] = {};
    compute_step(lA,        lB,        wave, lane, acc);
    compute_step(lA + 4096, lB + 2048, wave, lane, acc);

    const float scale = 0.125f; // 1/sqrt(64)
    int g  = (lane >> 4) & 1;
    int nc = lane & 15;
    int q0 = qB + wave * 16;
    float* Lrow = Lg + (size_t)bh * SS * SS;
#pragma unroll
    for (int t = 0; t < 4; ++t) {
#pragma unroll
        for (int r = 0; r < 8; ++r) {
            int q = q0 + r + 8 * g;
            int n = n0 + t * 16 + nc;
            float v = acc[t][r] * scale;
            v += (float)mask[((size_t)b * SS + q) * SS + n] * (-1e9f);
            Lrow[(size_t)q * SS + n] = v;
        }
    }
}

// ---------------- row softmax over 2048 elems, in place ----------------
__global__ __launch_bounds__(256)
void softmax2048_kernel(float* __restrict__ P) {
    __shared__ float red[256];
    int tid = threadIdx.x;
    float* p = P + (size_t)blockIdx.x * SS;

    float vals[8];
    float m = -3.4e38f;
#pragma unroll
    for (int i = 0; i < 8; ++i) {
        vals[i] = p[i * 256 + tid];
        m = fmaxf(m, vals[i]);
    }
    red[tid] = m; __syncthreads();
    for (int off = 128; off > 0; off >>= 1) {
        if (tid < off) red[tid] = fmaxf(red[tid], red[tid + off]);
        __syncthreads();
    }
    m = red[0]; __syncthreads();

    float s = 0.0f;
#pragma unroll
    for (int i = 0; i < 8; ++i) {
        vals[i] = __expf(vals[i] - m);
        s += vals[i];
    }
    red[tid] = s; __syncthreads();
    for (int off = 128; off > 0; off >>= 1) {
        if (tid < off) red[tid] += red[tid + off];
        __syncthreads();
    }
    float inv = 1.0f / red[0];
#pragma unroll
    for (int i = 0; i < 8; ++i) p[i * 256 + tid] = vals[i] * inv;
}

// ---------------- PV: ctx = P @ V, concat layout [B*S, H*DH] ----------------
__global__ __launch_bounds__(256)
void attn_pv_kernel(const float* __restrict__ P, const float* __restrict__ V,
                    float* __restrict__ CTX) {
    __shared__ _Float16 lA[2 * 128 * 32];
    __shared__ _Float16 lB[2 * 4 * 512];
    int tid  = threadIdx.x;
    int lane = tid & 31;
    int wave = tid >> 5;
    int bh = blockIdx.y;
    int b = bh / HH, h = bh % HH;
    int qB = blockIdx.x * 128;

    const float* Ph = P + (size_t)bh * SS * SS;                  // [S,S]
    const float* Vh = V + (size_t)b * SS * (HH * DHD) + h * DHD; // [S,64], ld=1024

    RegA ra = gload_a(Ph, SS, qB, 0, tid);
    RegB rb = gload_b(Vh, HH * DHD, 0, 0, tid);
    sstore_a(lA, ra, tid);
    sstore_b(lB, rb, tid);

    v8f acc[4] = {};
    int cur = 0;
    for (int k0 = 0; k0 < SS; k0 += 32) {
        __syncthreads();
        bool more = (k0 + 32) < SS;
        if (more) {
            ra = gload_a(Ph, SS, qB, k0 + 32, tid);
            rb = gload_b(Vh, HH * DHD, k0 + 32, 0, tid);
        }
        compute_step(lA + cur * 4096, lB + cur * 2048, wave, lane, acc);
        if (more) {
            sstore_a(lA + (cur ^ 1) * 4096, ra, tid);
            sstore_b(lB + (cur ^ 1) * 2048, rb, tid);
            cur ^= 1;
        }
    }

    int g  = (lane >> 4) & 1;
    int nc = lane & 15;
    int q0 = qB + wave * 16;
#pragma unroll
    for (int t = 0; t < 4; ++t) {
#pragma unroll
        for (int r = 0; r < 8; ++r) {
            int q = q0 + r + 8 * g;
            int d = t * 16 + nc;
            CTX[((size_t)b * SS + q) * (HH * DHD) + h * DHD + d] = acc[t][r];
        }
    }
}

// ---------------- LayerNorm over rows of D=1024 ----------------
__global__ __launch_bounds__(256)
void layernorm_kernel(const float* __restrict__ X, const float* __restrict__ gamma,
                      const float* __restrict__ beta, float* __restrict__ Y) {
    __shared__ float red[256];
    int tid = threadIdx.x;
    const float* x = X + (size_t)blockIdx.x * DD;
    float* y = Y + (size_t)blockIdx.x * DD;

    float local[4];
    float s = 0.0f;
#pragma unroll
    for (int i = 0; i < 4; ++i) {
        local[i] = x[i * 256 + tid];
        s += local[i];
    }
    red[tid] = s; __syncthreads();
    for (int off = 128; off > 0; off >>= 1) {
        if (tid < off) red[tid] += red[tid + off];
        __syncthreads();
    }
    float mu = red[0] * (1.0f / DD); __syncthreads();

    float vs = 0.0f;
#pragma unroll
    for (int i = 0; i < 4; ++i) {
        float d = local[i] - mu;
        vs += d * d;
    }
    red[tid] = vs; __syncthreads();
    for (int off = 128; off > 0; off >>= 1) {
        if (tid < off) red[tid] += red[tid + off];
        __syncthreads();
    }
    float rstd = rsqrtf(red[0] * (1.0f / DD) + 1e-5f); __syncthreads();

#pragma unroll
    for (int i = 0; i < 4; ++i) {
        int col = i * 256 + tid;
        y[col] = (local[i] - mu) * rstd * gamma[col] + beta[col];
    }
}

// ---------------- workspace layout (float offsets) ----------------
static const size_t ROWS   = (size_t)BB * SS;       // 4096
static const size_t OFF_Q  = 0;
static const size_t OFF_K  = OFF_Q  + ROWS * DD;
static const size_t OFF_V  = OFF_K  + ROWS * DD;
static const size_t OFF_CT = OFF_V  + ROWS * DD;
static const size_t OFF_Y1 = OFF_CT + ROWS * DD;
static const size_t OFF_X1 = OFF_Y1 + ROWS * DD;
static const size_t OFF_HB = OFF_X1 + ROWS * DD;
static const size_t OFF_Y2 = OFF_HB + ROWS * PFF;   // total ~46.1M floats ~176MB

extern "C" void kernel_launch(void* const* d_in, const int* in_sizes, int n_in,
                              void* d_out, int out_size, void* d_ws, size_t ws_size,
                              hipStream_t stream) {
    const float* inputs = (const float*)d_in[0];
    const int*   mask   = (const int*)  d_in[1];
    const float* Wq = (const float*)d_in[2];  const float* bq  = (const float*)d_in[3];
    const float* Wk = (const float*)d_in[4];  const float* bk  = (const float*)d_in[5];
    const float* Wv = (const float*)d_in[6];  const float* bv  = (const float*)d_in[7];
    const float* Wo = (const float*)d_in[8];  const float* bo  = (const float*)d_in[9];
    const float* W1 = (const float*)d_in[10]; const float* bf1 = (const float*)d_in[11];
    const float* W2 = (const float*)d_in[12]; const float* bf2 = (const float*)d_in[13];
    const float* ln1g = (const float*)d_in[14]; const float* ln1b = (const float*)d_in[15];
    const float* ln2g = (const float*)d_in[16]; const float* ln2b = (const float*)d_in[17];

    float* ws   = (float*)d_ws;
    float* outp = (float*)d_out;                       // [B,S,D]
    float* attn = outp + (size_t)BB * SS * DD;         // [B,H,S,S]

    float* Q  = ws + OFF_Q;
    float* Kb = ws + OFF_K;
    float* V  = ws + OFF_V;
    float* CT = ws + OFF_CT;
    float* Y1 = ws + OFF_Y1;
    float* X1 = ws + OFF_X1;
    float* HB = ws + OFF_HB;
    float* Y2 = ws + OFF_Y2;

    const int M = (int)ROWS;   // 4096
    dim3 blk(256);

    // 1) QKV projections: [4096,1024] @ [1024,1024]
    dim3 gProj(DD / 64, M / 128);
    gemm_wmma_kernel<0,0><<<gProj, blk, 0, stream>>>(inputs, Wq, bq, nullptr, Q,  M, DD, DD);
    gemm_wmma_kernel<0,0><<<gProj, blk, 0, stream>>>(inputs, Wk, bk, nullptr, Kb, M, DD, DD);
    gemm_wmma_kernel<0,0><<<gProj, blk, 0, stream>>>(inputs, Wv, bv, nullptr, V,  M, DD, DD);

    // 2) logits -> d_out attn region
    dim3 gLog(SS / 64, SS / 128, BB * HH);
    attn_logits_kernel<<<gLog, blk, 0, stream>>>(Q, Kb, mask, attn);

    // 3) softmax rows in place (B*H*S rows of length S)
    softmax2048_kernel<<<dim3(BB * HH * SS), blk, 0, stream>>>(attn);

    // 4) ctx = P @ V
    dim3 gPV(SS / 128, BB * HH);
    attn_pv_kernel<<<gPV, blk, 0, stream>>>(attn, V, CT);

    // 5) out projection + residual -> Y1; LN1 -> X1
    gemm_wmma_kernel<0,1><<<gProj, blk, 0, stream>>>(CT, Wo, bo, inputs, Y1, M, DD, DD);
    layernorm_kernel<<<dim3(M), blk, 0, stream>>>(Y1, ln1g, ln1b, X1);

    // 6) FFN1 (ReLU): [4096,1024]@[1024,4096]
    dim3 gF1(PFF / 64, M / 128);
    gemm_wmma_kernel<1,0><<<gF1, blk, 0, stream>>>(X1, W1, bf1, nullptr, HB, M, PFF, DD);

    // 7) FFN2 + residual(X1) -> Y2; LN2 -> d_out
    gemm_wmma_kernel<0,1><<<gProj, blk, 0, stream>>>(HB, W2, bf2, X1, Y2, M, DD, PFF);
    layernorm_kernel<<<dim3(M), blk, 0, stream>>>(Y2, ln2g, ln2b, outp);
}
